// PolyhedronRegressionModel_12326556140167
// MI455X (gfx1250) — matrix operand order repaired and checked by reading, not compiled, not run on hardware
//
#include <hip/hip_runtime.h>
#include <hip/hip_bf16.h>
#include <stdint.h>

// ---------------------------------------------------------------------------
// CDNA5 (gfx1250): CGConv x2 -> node MLP -> pool -> graph MLP
// v_wmma_f32_16x16x32_bf16, LDS-staged weights (async global->LDS),
// persistent workgroups for weight reuse. LDS B reads pinned in-loop to
// prevent LICM register-spill explosion (seen in round 3 asm).
// ---------------------------------------------------------------------------

typedef __attribute__((ext_vector_type(16))) __bf16 v16bf;
typedef __attribute__((ext_vector_type(8)))  float  v8f;
typedef __attribute__((ext_vector_type(4)))  int    v4i;

typedef __attribute__((address_space(3))) v16bf lds_v16bf;
typedef __attribute__((address_space(3))) char  lds_char;
typedef __attribute__((address_space(1))) v4i   g_v4i;   // global 16B chunk
typedef __attribute__((address_space(3))) v4i   l_v4i;   // LDS 16B chunk

#define N_NODES  50000
#define N_EDGES  800000
#define FDIM     128
#define EDIM     16
#define NGRAPH   512
#define ZIN      272     // 2F + D
#define KT_EDGE  9       // ceil(272/32) k-tiles (padded to 288)
#define NT_FS    16      // 256 fused output cols (f | s)
#define EDGE_TPB 256     // 8 waves
#define NODE_TPB 256

// ---------------------------------------------------------------------------
// Weight packing: B-fragment striping for V_WMMA_*_16X16X32 (bf16).
// Tile (kt, nt) holds B[kt*32 .. kt*32+31][nt*16 .. nt*16+15].
// lane = ((k%16) / 8)*16 + (n%16), elem = (k%32 / 16)*8 + k%8.
// ---------------------------------------------------------------------------
__global__ void pack_b_kernel(__bf16* __restrict__ dst,
                              const float* __restrict__ A,
                              const float* __restrict__ B,   // second half (or null)
                              int K, int N1, int KT, int NT) {
    int t = blockIdx.x * blockDim.x + threadIdx.x;
    int total = KT * NT * 512;
    if (t >= total) return;
    int tile   = t >> 9;
    int within = t & 511;
    int lane   = within >> 4;
    int elem   = within & 15;
    int kt = tile / NT, nt = tile % NT;
    int k = kt * 32 + (elem >> 3) * 16 + (lane >> 4) * 8 + (elem & 7);
    int n = nt * 16 + (lane & 15);
    float v = 0.f;
    if (k < K) {
        if (n < N1)      v = A[(int64_t)k * N1 + n];
        else if (B)      v = B[(int64_t)k * N1 + (n - N1)];
    }
    dst[t] = (__bf16)v;
}

// Load 8 consecutive f32 and convert to 8 bf16 (two 16B global loads).
__device__ __forceinline__ void cvt8(const float* __restrict__ p, __bf16* o) {
    float4 u = ((const float4*)p)[0];
    float4 v = ((const float4*)p)[1];
    o[0] = (__bf16)u.x; o[1] = (__bf16)u.y; o[2] = (__bf16)u.z; o[3] = (__bf16)u.w;
    o[4] = (__bf16)v.x; o[5] = (__bf16)v.y; o[6] = (__bf16)v.z; o[7] = (__bf16)v.w;
}

// Cooperative stage of packed weights into LDS via the CDNA5 async
// global->LDS data path (falls back to a VGPR round-trip if unavailable).
__device__ __forceinline__ void stage_to_lds(char* smem, const __bf16* Wt, int nbytes) {
#if __has_builtin(__builtin_amdgcn_global_load_async_to_lds_b128) && \
    __has_builtin(__builtin_amdgcn_s_wait_asynccnt)
    g_v4i* g = (g_v4i*)Wt;               // global 16B-chunk pointer (AS1)
    l_v4i* l = (l_v4i*)(lds_char*)smem;  // LDS 16B-chunk pointer (AS3)
    const int nchunks = nbytes >> 4;
    for (int i = threadIdx.x; i < nchunks; i += blockDim.x)
        __builtin_amdgcn_global_load_async_to_lds_b128(g + i, l + i, 0, 0);
    __builtin_amdgcn_s_wait_asynccnt(0);
#else
    const float4* src = (const float4*)Wt;
    float4*       dst = (float4*)smem;
    for (int i = threadIdx.x; i < (nbytes >> 4); i += blockDim.x)
        dst[i] = src[i];
#endif
    __syncthreads();
}

// ---------------------------------------------------------------------------
// CGConv edge kernel (persistent): one wave per 16-edge tile per iteration.
//   Z[16 x 288] (gathered bf16) @ Wcat[288 x 256] -> C[16 x 256] f32
//   msg = sigmoid(C_f + bf) * softplus(C_s + bs); atomic scatter to agg[dst].
// ---------------------------------------------------------------------------
__global__ void __launch_bounds__(EDGE_TPB)
cgconv_edge_kernel(const float*   __restrict__ x,
                   const float*   __restrict__ edge_attr,
                   const int64_t* __restrict__ edge_index,   // [2][E]
                   const __bf16*  __restrict__ Wt,           // packed [9][16] tiles
                   const float*   __restrict__ bf,
                   const float*   __restrict__ bs,
                   float*         __restrict__ agg) {
    extern __shared__ char smem[];
    constexpr int WBYTES = KT_EDGE * NT_FS * 512 * 2;        // 147456
    stage_to_lds(smem, Wt, WBYTES);

    const int wave = threadIdx.x >> 5;
    const int lane = threadIdx.x & 31;
    const int half = lane >> 4;
    const int m    = lane & 15;
    const int64_t* srcp = edge_index;
    const int64_t* dstp = edge_index + N_EDGES;
    const lds_v16bf* Bl = ((const lds_v16bf*)(lds_char*)smem) + lane;

    // per-lane bias values (col = nt*16 + lane%16), hoisted out of the loop
    float bFv[8], bSv[8];
    #pragma unroll
    for (int nt = 0; nt < 8; ++nt) {
        bFv[nt] = bf[nt * 16 + m];
        bSv[nt] = bs[nt * 16 + m];
    }

    const int nTiles     = N_EDGES / 16;
    const int wavesTotal = gridDim.x * (EDGE_TPB / 32);

    #pragma clang loop unroll(disable)
    for (int tile = blockIdx.x * (EDGE_TPB / 32) + wave; tile < nTiles; tile += wavesTotal) {
        // Prevent hoisting of the (loop-invariant) LDS weight loads out of this
        // loop: without this, LICM preloads ~144 B-fragments and spills to
        // scratch (observed in round-3 asm). Forces ds_load_b128 per use.
        asm volatile("" ::: "memory");

        const int base = tile * 16;
        const int64_t e  = base + m;
        const int64_t sN = srcp[e];
        const int64_t dN = dstp[e];
        const float* xs = x + sN * FDIM;
        const float* xd = x + dN * FDIM;
        const float* ea = edge_attr + e * EDIM;

        v8f c[16] = {};

        #pragma unroll
        for (int kt = 0; kt < KT_EDGE; ++kt) {
            union { v16bf v; __bf16 h[16]; } a;
            #pragma unroll
            for (int r = 0; r < 2; ++r) {
                const int kb = kt * 32 + r * 16;    // compile-time after unroll
                const int ks = kb + half * 8;
                __bf16* o = a.h + r * 8;
                if (kb < 128) {
                    cvt8(xd + ks, o);
                } else if (kb < 256) {
                    cvt8(xs + (ks - 128), o);
                } else if (kb < ZIN) {
                    cvt8(ea + (ks - 256), o);
                } else {
                    #pragma unroll
                    for (int j = 0; j < 8; ++j) o[j] = (__bf16)0.f;
                }
            }
            #pragma unroll
            for (int nt = 0; nt < NT_FS; ++nt) {
                v16bf b = Bl[(kt * NT_FS + nt) * 32];          // ds_load_b128 x2
                c[nt] = __builtin_amdgcn_wmma_f32_16x16x32_bf16(
                            false, a.v, false, b, (short)0, c[nt], false, false);
            }
        }

        // C layout: VGPR j of lane l holds row M = j + 8*(l>>4), col nt*16+(l&15)
        int64_t drow[8];
        #pragma unroll
        for (int j = 0; j < 8; ++j) drow[j] = dstp[base + j + 8 * half];

        #pragma unroll
        for (int nt = 0; nt < 8; ++nt) {
            const int f = nt * 16 + m;
            #pragma unroll
            for (int j = 0; j < 8; ++j) {
                float vf = c[nt][j]     + bFv[nt];
                float vs = c[nt + 8][j] + bSv[nt];
                float gate = 1.f / (1.f + __expf(-vf));
                float sp   = (vs > 20.f) ? vs : __logf(1.f + __expf(vs));
                atomicAdd(&agg[drow[j] * FDIM + f], gate * sp);
            }
        }
    }
}

// h = a + b (residual add), n elements
__global__ void add_kernel(const float* __restrict__ a,
                           const float* __restrict__ b,
                           float* __restrict__ o, int64_t n) {
    int64_t i = (int64_t)blockIdx.x * blockDim.x + threadIdx.x;
    if (i < n) o[i] = a[i] + b[i];
}

// ---------------------------------------------------------------------------
// Node GEMM (rows x K) @ (K x 256) + bias, ReLU. If pool != null, atomically
// accumulate into g[pool[row]*256 + col]. Persistent blocks, weights in LDS.
// ---------------------------------------------------------------------------
__global__ void __launch_bounds__(NODE_TPB)
node_gemm_kernel(const float*   __restrict__ A, int K, int KT,
                 const __bf16*  __restrict__ Wt,
                 const float*   __restrict__ bias,
                 float*         __restrict__ out,
                 const int64_t* __restrict__ pool,
                 float*         __restrict__ g,
                 int rows) {
    extern __shared__ char smem[];
    stage_to_lds(smem, Wt, KT * 16 * 512 * 2);

    const int wave = threadIdx.x >> 5;
    const int lane = threadIdx.x & 31;
    const int half = lane >> 4;
    const int m    = lane & 15;
    const lds_v16bf* Bl = ((const lds_v16bf*)(lds_char*)smem) + lane;

    float bv[16];
    #pragma unroll
    for (int nt = 0; nt < 16; ++nt) bv[nt] = bias[nt * 16 + m];

    const int nTiles     = rows / 16;
    const int wavesTotal = gridDim.x * (NODE_TPB / 32);

    #pragma clang loop unroll(disable)
    for (int tile = blockIdx.x * (NODE_TPB / 32) + wave; tile < nTiles; tile += wavesTotal) {
        asm volatile("" ::: "memory");   // keep LDS B loads inside the loop

        const int base = tile * 16;
        const float* Ar = A + (int64_t)(base + m) * K;

        v8f c[16] = {};
        #pragma clang loop unroll(disable)
        for (int kt = 0; kt < KT; ++kt) {
            union { v16bf v; __bf16 h[16]; } a;
            #pragma unroll
            for (int r = 0; r < 2; ++r) {
                const int ks = kt * 32 + r * 16 + half * 8;
                cvt8(Ar + ks, a.h + r * 8);
            }
            #pragma unroll
            for (int nt = 0; nt < 16; ++nt) {
                v16bf b = Bl[(kt * 16 + nt) * 32];
                c[nt] = __builtin_amdgcn_wmma_f32_16x16x32_bf16(
                            false, a.v, false, b, (short)0, c[nt], false, false);
            }
        }

        #pragma unroll
        for (int nt = 0; nt < 16; ++nt) {
            const int f = nt * 16 + m;
            #pragma unroll
            for (int j = 0; j < 8; ++j) {
                const int row = base + j + 8 * half;
                float v = c[nt][j] + bv[nt];
                v = v > 0.f ? v : 0.f;
                if (pool) atomicAdd(&g[pool[row] * 256 + f], v);
                else      out[(int64_t)row * 256 + f] = v;
            }
        }
    }
}

// g2 = relu(g @ W2a + b2a): [512,256] @ [256,128]
__global__ void graph_mlp_kernel(const float* __restrict__ g,
                                 const float* __restrict__ W2a,
                                 const float* __restrict__ b2a,
                                 float* __restrict__ g2) {
    int t = blockIdx.x * blockDim.x + threadIdx.x;
    if (t >= NGRAPH * 128) return;
    int gi = t >> 7, col = t & 127;
    const float* gr = g + gi * 256;
    float acc = b2a[col];
    for (int k = 0; k < 256; ++k) acc = fmaf(gr[k], W2a[k * 128 + col], acc);
    g2[t] = acc > 0.f ? acc : 0.f;
}

// out = g2 @ Wo + bo : [512,128] @ [128,1]
__global__ void out_kernel(const float* __restrict__ g2,
                           const float* __restrict__ Wo,
                           const float* __restrict__ bo,
                           float* __restrict__ out) {
    int gi = blockIdx.x * blockDim.x + threadIdx.x;
    if (gi >= NGRAPH) return;
    const float* gr = g2 + gi * 128;
    float acc = bo[0];
    for (int k = 0; k < 128; ++k) acc = fmaf(gr[k], Wo[k], acc);
    out[gi] = acc;
}

// ---------------------------------------------------------------------------
extern "C" void kernel_launch(void* const* d_in, const int* in_sizes, int n_in,
                              void* d_out, int out_size, void* d_ws, size_t ws_size,
                              hipStream_t stream) {
    const float*   x          = (const float*)  d_in[0];
    const float*   edge_attr  = (const float*)  d_in[1];
    const int64_t* edge_index = (const int64_t*)d_in[2];
    const int64_t* batch      = (const int64_t*)d_in[3];
    const float* Wf1 = (const float*)d_in[4];  const float* bf1 = (const float*)d_in[5];
    const float* Ws1 = (const float*)d_in[6];  const float* bs1 = (const float*)d_in[7];
    const float* Wf2 = (const float*)d_in[8];  const float* bf2 = (const float*)d_in[9];
    const float* Ws2 = (const float*)d_in[10]; const float* bs2 = (const float*)d_in[11];
    const float* W1a = (const float*)d_in[12]; const float* b1a = (const float*)d_in[13];
    const float* W1b = (const float*)d_in[14]; const float* b1b = (const float*)d_in[15];
    const float* W2a = (const float*)d_in[16]; const float* b2a = (const float*)d_in[17];
    const float* Wo  = (const float*)d_in[18]; const float* bo  = (const float*)d_in[19];

    // ---- workspace carve-up (256B aligned) ----
    char* p = (char*)d_ws;
    auto carve = [&](size_t bytes) -> char* {
        char* r = p;
        p += (bytes + 255) & ~(size_t)255;
        return r;
    };
    __bf16* Wc1  = (__bf16*)carve(KT_EDGE * NT_FS * 512 * sizeof(__bf16)); // 288x256
    __bf16* Wc2  = (__bf16*)carve(KT_EDGE * NT_FS * 512 * sizeof(__bf16));
    __bf16* W1ap = (__bf16*)carve(4 * 16 * 512 * sizeof(__bf16));          // 128x256
    __bf16* W1bp = (__bf16*)carve(8 * 16 * 512 * sizeof(__bf16));          // 256x256
    float*  agg  = (float*)carve((size_t)N_NODES * FDIM * sizeof(float));
    float*  h    = (float*)carve((size_t)N_NODES * FDIM * sizeof(float));
    float*  h1   = (float*)carve((size_t)N_NODES * 256  * sizeof(float));
    float*  gbuf = (float*)carve((size_t)NGRAPH * 256   * sizeof(float));
    float*  g2   = (float*)carve((size_t)NGRAPH * 128   * sizeof(float));

    const int edgeLds  = KT_EDGE * NT_FS * 512 * 2;   // 147456 B
    const int nodeLds1 = 4 * 16 * 512 * 2;            // 65536 B
    const int nodeLds2 = 8 * 16 * 512 * 2;            // 131072 B
    (void)hipFuncSetAttribute((const void*)cgconv_edge_kernel,
                              hipFuncAttributeMaxDynamicSharedMemorySize, edgeLds);
    (void)hipFuncSetAttribute((const void*)node_gemm_kernel,
                              hipFuncAttributeMaxDynamicSharedMemorySize, nodeLds2);

    // ---- zero the accumulation buffers ----
    (void)hipMemsetAsync(agg,  0, (size_t)N_NODES * FDIM * sizeof(float), stream);
    (void)hipMemsetAsync(gbuf, 0, (size_t)NGRAPH * 256   * sizeof(float), stream);

    // ---- pack weights to bf16 WMMA B-fragments ----
    {
        int tot = KT_EDGE * NT_FS * 512;
        pack_b_kernel<<<(tot + 255) / 256, 256, 0, stream>>>(Wc1, Wf1, Ws1, ZIN, 128, KT_EDGE, NT_FS);
        pack_b_kernel<<<(tot + 255) / 256, 256, 0, stream>>>(Wc2, Wf2, Ws2, ZIN, 128, KT_EDGE, NT_FS);
        int t1 = 4 * 16 * 512;
        pack_b_kernel<<<(t1 + 255) / 256, 256, 0, stream>>>(W1ap, W1a, nullptr, 128, 256, 4, 16);
        int t2 = 8 * 16 * 512;
        pack_b_kernel<<<(t2 + 255) / 256, 256, 0, stream>>>(W1bp, W1b, nullptr, 256, 256, 8, 16);
    }

    const int edgeBlocks = 800;   // persistent; ~7.8 tiles per wave
    const int nodeBlocks = 400;   // persistent; 3125 tiles total

    // ---- CGConv layer 1 ----
    cgconv_edge_kernel<<<edgeBlocks, EDGE_TPB, edgeLds, stream>>>(
        x, edge_attr, edge_index, Wc1, bf1, bs1, agg);
    {
        int64_t n = (int64_t)N_NODES * FDIM;
        add_kernel<<<(int)((n + 255) / 256), 256, 0, stream>>>(x, agg, h, n);
    }

    // ---- CGConv layer 2 ----
    (void)hipMemsetAsync(agg, 0, (size_t)N_NODES * FDIM * sizeof(float), stream);
    cgconv_edge_kernel<<<edgeBlocks, EDGE_TPB, edgeLds, stream>>>(
        h, edge_attr, edge_index, Wc2, bf2, bs2, agg);
    {
        int64_t n = (int64_t)N_NODES * FDIM;
        add_kernel<<<(int)((n + 255) / 256), 256, 0, stream>>>(h, agg, h, n);
    }

    // ---- node MLP ----
    node_gemm_kernel<<<nodeBlocks, NODE_TPB, nodeLds1, stream>>>(
        h,  128, 4, W1ap, b1a, h1, nullptr, nullptr, N_NODES);
    // second layer fused with global-add-pool
    node_gemm_kernel<<<nodeBlocks, NODE_TPB, nodeLds2, stream>>>(
        h1, 256, 8, W1bp, b1b, nullptr, batch, gbuf, N_NODES);

    // ---- graph head ----
    graph_mlp_kernel<<<(NGRAPH * 128 + 255) / 256, 256, 0, stream>>>(gbuf, W2a, b2a, g2);
    out_kernel<<<(NGRAPH + 255) / 256, 256, 0, stream>>>(g2, Wo, bo, (float*)d_out);
}
